// Sequence_Modeling_28991029248622
// MI455X (gfx1250) — matrix operand order repaired and checked by reading, not compiled, not run on hardware
//
#include <hip/hip_runtime.h>
#include <hip/hip_bf16.h>

#define DEVI __device__ __forceinline__

typedef __attribute__((ext_vector_type(16))) __bf16 v16bf;
typedef __attribute__((ext_vector_type(8)))  float  v8f;

union Frag { v16bf v; uint4 q[2]; };

// ---- bf16 <-> f32 via bit ops (round-to-nearest-even) ----
DEVI __bf16 f2bf(float x) {
  union { float f; unsigned u; } v; v.f = x;
  unsigned r = v.u + 0x7FFFu + ((v.u >> 16) & 1u);
  union { unsigned short s; __bf16 b; } o; o.s = (unsigned short)(r >> 16);
  return o.b;
}

// ---- WMMA fragment loaders (layouts per CDNA5 ISA §7.12.2) ----
// A (16x32 bf16): lane l in [0,16): row M=l, K = {0..7, 16..23}; lane l+16: K = {8..15, 24..31}
DEVI v16bf load_a(const __bf16* A, int lda, int k0, int lane) {
  int l = lane & 15, hi = lane >> 4;
  const __bf16* p = A + (size_t)l * lda + k0 + hi * 8;
  Frag f;
  f.q[0] = *(const uint4*)(p);
  f.q[1] = *(const uint4*)(p + 16);
  return f.v;
}
// B (32x16 bf16) for C = A @ W^T: column n = row n of W. lane n: K = {0..7, 16..23}; lane n+16: K = {8..15, 24..31}
DEVI v16bf load_b(const __bf16* W, int ldw, int k0, int lane) {
  int n = lane & 15, hi = lane >> 4;
  const __bf16* p = W + (size_t)n * ldw + k0 + hi * 8;
  Frag f;
  f.q[0] = *(const uint4*)(p);
  f.q[1] = *(const uint4*)(p + 16);
  return f.v;
}
// A fragment where only row M=0 is valid (vector attn row); other rows zero
DEVI v16bf load_row0(const __bf16* a, int k0, int lane) {
  int l = lane & 15, hi = lane >> 4;
  Frag f;
  if (l == 0) {
    const __bf16* p = a + k0 + hi * 8;
    f.q[0] = *(const uint4*)(p);
    f.q[1] = *(const uint4*)(p + 16);
  } else {
    f.q[0] = make_uint4(0u, 0u, 0u, 0u);
    f.q[1] = make_uint4(0u, 0u, 0u, 0u);
  }
  return f.v;
}

DEVI v8f wmma_bf16(v16bf a, v16bf b, v8f c) {
  return __builtin_amdgcn_wmma_f32_16x16x32_bf16(false, a, false, b, (short)0, c, false, false);
}

// ---- Rotating two-buffer K pipelines (K must be a multiple of 64).
// Each load writes directly into the register set just consumed by a WMMA, so the
// rotation needs no register moves, and loads lead their consumer by two K-steps. ----
DEVI void mm1(const __bf16* A, int lda, const __bf16* W, int ldw, int K, int lane, v8f& acc) {
  v16bf a0 = load_a(A, lda, 0, lane),  b0 = load_b(W, ldw, 0, lane);
  v16bf a1 = load_a(A, lda, 32, lane), b1 = load_b(W, ldw, 32, lane);
  for (int k0 = 64; k0 < K; k0 += 64) {
    acc = wmma_bf16(a0, b0, acc);
    a0 = load_a(A, lda, k0, lane); b0 = load_b(W, ldw, k0, lane);
    acc = wmma_bf16(a1, b1, acc);
    a1 = load_a(A, lda, k0 + 32, lane); b1 = load_b(W, ldw, k0 + 32, lane);
  }
  acc = wmma_bf16(a0, b0, acc);
  acc = wmma_bf16(a1, b1, acc);
}

DEVI void mm2(const __bf16* A, int lda, const __bf16* W0, const __bf16* W1, int ldw,
              int K, int lane, v8f& acc0, v8f& acc1) {
  v16bf a0  = load_a(A, lda, 0, lane);
  v16bf p0  = load_b(W0, ldw, 0, lane),  q0 = load_b(W1, ldw, 0, lane);
  v16bf a1  = load_a(A, lda, 32, lane);
  v16bf p1  = load_b(W0, ldw, 32, lane), q1 = load_b(W1, ldw, 32, lane);
  for (int k0 = 64; k0 < K; k0 += 64) {
    acc0 = wmma_bf16(a0, p0, acc0);
    acc1 = wmma_bf16(a0, q0, acc1);
    a0 = load_a(A, lda, k0, lane);
    p0 = load_b(W0, ldw, k0, lane); q0 = load_b(W1, ldw, k0, lane);
    acc0 = wmma_bf16(a1, p1, acc0);
    acc1 = wmma_bf16(a1, q1, acc1);
    a1 = load_a(A, lda, k0 + 32, lane);
    p1 = load_b(W0, ldw, k0 + 32, lane); q1 = load_b(W1, ldw, k0 + 32, lane);
  }
  acc0 = wmma_bf16(a0, p0, acc0);
  acc1 = wmma_bf16(a0, q0, acc1);
  acc0 = wmma_bf16(a1, p1, acc0);
  acc1 = wmma_bf16(a1, q1, acc1);
}

DEVI void mm1_r0(const __bf16* arow, const __bf16* W, int ldw, int K, int lane, v8f& acc) {
  v16bf a0 = load_row0(arow, 0, lane),  b0 = load_b(W, ldw, 0, lane);
  v16bf a1 = load_row0(arow, 32, lane), b1 = load_b(W, ldw, 32, lane);
  for (int k0 = 64; k0 < K; k0 += 64) {
    acc = wmma_bf16(a0, b0, acc);
    a0 = load_row0(arow, k0, lane); b0 = load_b(W, ldw, k0, lane);
    acc = wmma_bf16(a1, b1, acc);
    a1 = load_row0(arow, k0 + 32, lane); b1 = load_b(W, ldw, k0 + 32, lane);
  }
  acc = wmma_bf16(a0, b0, acc);
  acc = wmma_bf16(a1, b1, acc);
}

// ---------------- elementwise helpers ----------------
__global__ void k_cvt(const float* __restrict__ src, __bf16* __restrict__ dst, size_t n) {
  size_t stride = (size_t)gridDim.x * blockDim.x;
  for (size_t i = (size_t)blockIdx.x * blockDim.x + threadIdx.x; i < n; i += stride)
    dst[i] = f2bf(src[i]);
}

// Gather embeddings: X[(t*16 + b)*512 + e] = emb[ids[b*S + t]*512 + e], as bf16
__global__ void k_embed(const int* __restrict__ ids, const float* __restrict__ emb,
                        __bf16* __restrict__ X, int S) {
  size_t total = (size_t)S * 16 * 512;
  size_t stride = (size_t)gridDim.x * blockDim.x;
  for (size_t i = (size_t)blockIdx.x * blockDim.x + threadIdx.x; i < total; i += stride) {
    int e   = (int)(i & 511);
    int row = (int)(i >> 9);
    int b = row & 15, tt = row >> 4;
    int id = ids[b * S + tt];
    X[i] = f2bf(emb[(size_t)id * 512 + e]);
  }
}

// ---------------- generic bf16 WMMA GEMM: C(MxN,f32) = A(MxK) @ W(NxK)^T + bias ----------------
// One wave computes a 32x64 tile (2 A-frags x 4 B-frags, 8 accumulators), with a rotating
// two-buffer K pipeline (zero-move) + global_prefetch of the streamed weight rows.
// permuteBT remaps output row m = t*16+b -> b*128+t (for the (B,SD,O) final output layout).
__global__ void __launch_bounds__(256)
k_gemm(const __bf16* __restrict__ A, int lda,
       const __bf16* __restrict__ W, int ldw, int wofs,
       const float* __restrict__ bias,
       float* __restrict__ C,
       int M, int N, int K, int permuteBT) {
  int wid  = (blockIdx.x * blockDim.x + threadIdx.x) >> 5;
  int lane = threadIdx.x & 31;
  int nt64 = N >> 6;
  int mt = wid / nt64;
  int nt = wid - mt * nt64;
  if (mt * 32 >= M) return;
  const __bf16* A0 = A + (size_t)(mt * 32) * lda;
  const __bf16* A1 = A0 + (size_t)16 * lda;
  int n0 = nt * 64;
  const __bf16* Wp[4];
#pragma unroll
  for (int j = 0; j < 4; ++j) Wp[j] = W + (size_t)(n0 + 16 * j) * ldw + wofs;

  v8f acc[8] = {v8f{}, v8f{}, v8f{}, v8f{}, v8f{}, v8f{}, v8f{}, v8f{}};
  // buffer 0 @ k=0, buffer 1 @ k=32
  v16bf xa0 = load_a(A0, lda, 0, lane), xa1 = load_a(A1, lda, 0, lane);
  v16bf ya0 = load_a(A0, lda, 32, lane), ya1 = load_a(A1, lda, 32, lane);
  v16bf xb[4], yb[4];
#pragma unroll
  for (int j = 0; j < 4; ++j) xb[j] = load_b(Wp[j], ldw, 0, lane);
#pragma unroll
  for (int j = 0; j < 4; ++j) yb[j] = load_b(Wp[j], ldw, 32, lane);

  size_t prow = (size_t)(lane & 15) * ldw;  // this lane's weight row
  for (int k0 = 64; k0 < K; k0 += 64) {
#pragma unroll
    for (int j = 0; j < 4; ++j)  // prefetch one step beyond the two-deep pipeline
      __builtin_prefetch(Wp[j] + prow + k0 + 64, 0, 1);
    // consume buffer 0 (k0-64), reload it at k0
#pragma unroll
    for (int j = 0; j < 4; ++j) acc[j] = wmma_bf16(xa0, xb[j], acc[j]);
#pragma unroll
    for (int j = 0; j < 4; ++j) acc[4 + j] = wmma_bf16(xa1, xb[j], acc[4 + j]);
    xa0 = load_a(A0, lda, k0, lane);
    xa1 = load_a(A1, lda, k0, lane);
#pragma unroll
    for (int j = 0; j < 4; ++j) xb[j] = load_b(Wp[j], ldw, k0, lane);
    // consume buffer 1 (k0-32), reload it at k0+32
#pragma unroll
    for (int j = 0; j < 4; ++j) acc[j] = wmma_bf16(ya0, yb[j], acc[j]);
#pragma unroll
    for (int j = 0; j < 4; ++j) acc[4 + j] = wmma_bf16(ya1, yb[j], acc[4 + j]);
    ya0 = load_a(A0, lda, k0 + 32, lane);
    ya1 = load_a(A1, lda, k0 + 32, lane);
#pragma unroll
    for (int j = 0; j < 4; ++j) yb[j] = load_b(Wp[j], ldw, k0 + 32, lane);
  }
#pragma unroll
  for (int j = 0; j < 4; ++j) acc[j] = wmma_bf16(xa0, xb[j], acc[j]);
#pragma unroll
  for (int j = 0; j < 4; ++j) acc[4 + j] = wmma_bf16(xa1, xb[j], acc[4 + j]);
#pragma unroll
  for (int j = 0; j < 4; ++j) acc[j] = wmma_bf16(ya0, yb[j], acc[j]);
#pragma unroll
  for (int j = 0; j < 4; ++j) acc[4 + j] = wmma_bf16(ya1, yb[j], acc[4 + j]);

  int nl = lane & 15, hi = lane >> 4;
#pragma unroll
  for (int half = 0; half < 2; ++half) {
#pragma unroll
    for (int j = 0; j < 4; ++j) {
      int n = n0 + 16 * j + nl;
      float bv = bias ? bias[n] : 0.0f;
#pragma unroll
      for (int r = 0; r < 8; ++r) {
        int m = mt * 32 + half * 16 + r + hi * 8;
        size_t row = permuteBT ? ((size_t)(m & 15) * 128 + (size_t)(m >> 4)) : (size_t)m;
        C[row * (size_t)N + n] = acc[half * 4 + j][r] + bv;
      }
    }
  }
}

// ---------------- encoder recurrence (persistent, 1 workgroup, 32 waves) ----------------
// h_t = tanh(Xpre_t + h_{t-1} @ Whh^T); Xpre has enc_b folded in.
// Stores hidden states in both (b,s,h) and (b,h,s) bf16 layouts for the decoder attention.
__global__ void __launch_bounds__(1024)
k_enc_rnn(const float* __restrict__ Xpre, const __bf16* __restrict__ Whh,
          const float* __restrict__ state0,
          __bf16* __restrict__ HencBS, __bf16* __restrict__ HencT) {
  extern __shared__ char dynsm[];
  __bf16* hsm = (__bf16*)dynsm;  // 16 x 1024 bf16
  int tid = threadIdx.x;
  for (int i = tid; i < 16 * 1024; i += 1024) hsm[i] = f2bf(state0[i]);
  __syncthreads();
  int wave = tid >> 5, lane = tid & 31;
  int l = lane & 15, hi = lane >> 4;
  int n0a = wave * 32;  // this wave owns N-tiles [n0a, n0a+32)
  for (int t = 0; t < 128; ++t) {
    v8f acc0 = {}, acc1 = {};
    const float* xp = Xpre + (size_t)t * 16 * 1024;
#pragma unroll
    for (int r = 0; r < 8; ++r) {
      acc0[r] = xp[(r + hi * 8) * 1024 + n0a + l];
      acc1[r] = xp[(r + hi * 8) * 1024 + n0a + 16 + l];
    }
    mm2(hsm, 1024, Whh + (size_t)n0a * 1024, Whh + (size_t)(n0a + 16) * 1024, 1024,
        1024, lane, acc0, acc1);
    __syncthreads();  // everyone done reading old h
#pragma unroll
    for (int r = 0; r < 8; ++r) {
      int b = r + hi * 8;
      {
        __bf16 bv = f2bf(tanhf(acc0[r]));
        int n = n0a + l;
        hsm[b * 1024 + n] = bv;
        HencBS[((size_t)b * 128 + t) * 1024 + n] = bv;
        HencT[((size_t)b * 1024 + n) * 128 + t] = bv;
      }
      {
        __bf16 bv = f2bf(tanhf(acc1[r]));
        int n = n0a + 16 + l;
        hsm[b * 1024 + n] = bv;
        HencBS[((size_t)b * 128 + t) * 1024 + n] = bv;
        HencT[((size_t)b * 1024 + n) * 128 + t] = bv;
      }
    }
    __syncthreads();  // new h visible
  }
}

// ---------------- decoder recurrence with attention (persistent, 1 workgroup) ----------------
__global__ void __launch_bounds__(1024)
k_dec_rnn(const float* __restrict__ Ypre,
          const __bf16* __restrict__ HencBS, const __bf16* __restrict__ HencT,
          const __bf16* __restrict__ W1, const float* __restrict__ b1,
          const __bf16* __restrict__ WihD, const __bf16* __restrict__ WhhD,
          __bf16* __restrict__ Hdec) {
  extern __shared__ char dynsm[];
  __bf16* hsm  = (__bf16*)dynsm;                      // 16x1024 h (32KB)
  __bf16* qcsm = (__bf16*)(dynsm + 32768);            // 16x1024 q, then reused for ctx (32KB)
  float*  ssm  = (float*)(dynsm + 65536);             // 16x128 scores f32 (8KB)
  __bf16* atn  = (__bf16*)(dynsm + 65536 + 8192);     // 16x128 attn bf16 (4KB)
  int tid = threadIdx.x, wave = tid >> 5, lane = tid & 31;
  int l = lane & 15, hi = lane >> 4;
  // h0 = encoder final state (t = 127)
  for (int i = tid; i < 16 * 1024; i += 1024)
    hsm[i] = HencBS[((size_t)(i >> 10) * 128 + 127) * 1024 + (i & 1023)];
  __syncthreads();
  for (int t = 0; t < 128; ++t) {
    // ---- q = h @ mlp1_W^T + mlp1_b ----
    for (int jt = wave; jt < 64; jt += 32) {
      int n0 = jt * 16;
      v8f acc = {};
      mm1(hsm, 1024, W1 + (size_t)n0 * 1024, 1024, 1024, lane, acc);
      float bb = b1[n0 + l];
#pragma unroll
      for (int r = 0; r < 8; ++r)
        qcsm[(r + hi * 8) * 1024 + n0 + l] = f2bf(acc[r] + bb);
    }
    __syncthreads();
    // ---- scores[b][s] = Henc_b(s,h) @ q_b : M = s-tile, B broadcast (ldw = 0) ----
    for (int j = wave; j < 128; j += 32) {
      int b = j >> 3, s0 = (j & 7) * 16;
      const __bf16* Ab = HencBS + ((size_t)b * 128 + s0) * 1024;
      const __bf16* qb = qcsm + b * 1024;
      v8f acc = {};
      mm1(Ab, 1024, qb, 0, 1024, lane, acc);  // all 16 B-columns identical
      if (l == 0) {
#pragma unroll
        for (int r = 0; r < 8; ++r) ssm[b * 128 + s0 + r + hi * 8] = acc[r];
      }
    }
    __syncthreads();
    // ---- softmax over s, one row per thread ----
    if (tid < 16) {
      float mx = -1e30f;
      for (int s = 0; s < 128; ++s) mx = fmaxf(mx, ssm[tid * 128 + s]);
      float sum = 0.f;
      for (int s = 0; s < 128; ++s) {
        float e = __expf(ssm[tid * 128 + s] - mx);
        ssm[tid * 128 + s] = e;
        sum += e;
      }
      float inv = 1.f / sum;
      for (int s = 0; s < 128; ++s) atn[tid * 128 + s] = f2bf(ssm[tid * 128 + s] * inv);
    }
    __syncthreads();
    // ---- ctx_b = attn_b @ Henc_b : A has one valid row, B from (b,h,s) layout ----
    for (int j = wave; j < 1024; j += 32) {
      int b = j >> 6, n0 = (j & 63) * 16;
      const __bf16* Bb = HencT + ((size_t)b * 1024 + n0) * 128;
      const __bf16* ab = atn + b * 128;
      v8f acc = {};
      mm1_r0(ab, Bb, 128, 128, lane, acc);
      if (hi == 0) qcsm[b * 1024 + n0 + l] = f2bf(acc[0]);  // row M=0 only
    }
    __syncthreads();
    // ---- h = tanh(Ypre_t + ctx @ WihR^T + h @ Whh^T) ----
    int n0a = wave * 32;
    v8f acc0 = {}, acc1 = {};
    const float* yp = Ypre + (size_t)t * 16 * 1024;
#pragma unroll
    for (int r = 0; r < 8; ++r) {
      acc0[r] = yp[(r + hi * 8) * 1024 + n0a + l];
      acc1[r] = yp[(r + hi * 8) * 1024 + n0a + 16 + l];
    }
    // ctx part: dec_Wih columns 512..1535
    mm2(qcsm, 1024, WihD + (size_t)n0a * 1536 + 512, WihD + (size_t)(n0a + 16) * 1536 + 512,
        1536, 1024, lane, acc0, acc1);
    // recurrent part
    mm2(hsm, 1024, WhhD + (size_t)n0a * 1024, WhhD + (size_t)(n0a + 16) * 1024,
        1024, 1024, lane, acc0, acc1);
    __syncthreads();  // done reading old h
#pragma unroll
    for (int r = 0; r < 8; ++r) {
      int b = r + hi * 8;
      {
        __bf16 bv = f2bf(tanhf(acc0[r]));
        int n = n0a + l;
        hsm[b * 1024 + n] = bv;
        Hdec[((size_t)t * 16 + b) * 1024 + n] = bv;
      }
      {
        __bf16 bv = f2bf(tanhf(acc1[r]));
        int n = n0a + 16 + l;
        hsm[b * 1024 + n] = bv;
        Hdec[((size_t)t * 16 + b) * 1024 + n] = bv;
      }
    }
    __syncthreads();  // new h visible
  }
}

// ---------------- host-side orchestration ----------------
extern "C" void kernel_launch(void* const* d_in, const int* in_sizes, int n_in,
                              void* d_out, int out_size, void* d_ws, size_t ws_size,
                              hipStream_t stream) {
  (void)in_sizes; (void)n_in; (void)out_size; (void)ws_size;
  const int*   enc_x  = (const int*)d_in[0];
  const int*   dec_y  = (const int*)d_in[1];
  const float* state0 = (const float*)d_in[2];
  const float* emb    = (const float*)d_in[3];
  const float* eWih   = (const float*)d_in[4];
  const float* eWhh   = (const float*)d_in[5];
  const float* enc_b  = (const float*)d_in[6];
  const float* m1W    = (const float*)d_in[7];
  const float* m1b    = (const float*)d_in[8];
  const float* dWih   = (const float*)d_in[9];
  const float* dWhh   = (const float*)d_in[10];
  const float* dec_b  = (const float*)d_in[11];
  const float* m2W    = (const float*)d_in[12];
  const float* m2b    = (const float*)d_in[13];
  float* out = (float*)d_out;

  char* ws = (char*)d_ws;
  __bf16* XE   = (__bf16*)(ws + 0);          // 2048 x 512
  __bf16* YD   = (__bf16*)(ws + 2097152);    // 2048 x 512
  __bf16* WIHE = (__bf16*)(ws + 4194304);    // 1024 x 512
  __bf16* WHHE = (__bf16*)(ws + 5242880);    // 1024 x 1024
  __bf16* W1B  = (__bf16*)(ws + 7340032);    // 1024 x 1024
  __bf16* WIHD = (__bf16*)(ws + 9437184);    // 1024 x 1536
  __bf16* WHHD = (__bf16*)(ws + 12582912);   // 1024 x 1024
  __bf16* W2B  = (__bf16*)(ws + 14680064);   // 32000 x 1024
  float*  XPRE = (float*)(ws + 80216064);    // 2048 x 1024
  float*  YPRE = (float*)(ws + 88604672);    // 2048 x 1024
  __bf16* HBS  = (__bf16*)(ws + 96993280);   // (b,s,h) 16x128x1024
  __bf16* HT   = (__bf16*)(ws + 101187584);  // (b,h,s) 16x1024x128
  __bf16* HDEC = (__bf16*)(ws + 105381888);  // (t*16+b, h) 2048x1024

  // weight conversions f32 -> bf16
  k_cvt<<<512, 256, 0, stream>>>(eWih, WIHE, (size_t)1024 * 512);
  k_cvt<<<512, 256, 0, stream>>>(eWhh, WHHE, (size_t)1024 * 1024);
  k_cvt<<<512, 256, 0, stream>>>(m1W, W1B, (size_t)1024 * 1024);
  k_cvt<<<512, 256, 0, stream>>>(dWih, WIHD, (size_t)1024 * 1536);
  k_cvt<<<512, 256, 0, stream>>>(dWhh, WHHD, (size_t)1024 * 1024);
  k_cvt<<<2048, 256, 0, stream>>>(m2W, W2B, (size_t)32000 * 1024);
  // embedding gathers
  k_embed<<<1024, 256, 0, stream>>>(enc_x, emb, XE, 128);
  k_embed<<<1024, 256, 0, stream>>>(dec_y, emb, YD, 128);
  // batched input projections (bias folded): M=2048, N=1024, K=512 -> (2048/32)*(1024/64)=1024 waves
  k_gemm<<<128, 256, 0, stream>>>(XE, 512, WIHE, 512, 0, enc_b, XPRE, 2048, 1024, 512, 0);
  k_gemm<<<128, 256, 0, stream>>>(YD, 512, WIHD, 1536, 0, dec_b, YPRE, 2048, 1024, 512, 0);
  // sequential scans (single WGP persistent kernels, LDS-resident hidden state)
  k_enc_rnn<<<1, 1024, 32768, stream>>>(XPRE, WHHE, state0, HBS, HT);
  k_dec_rnn<<<1, 1024, 77824, stream>>>(YPRE, HBS, HT, W1B, m1b, WIHD, WHHD, HDEC);
  // hoisted output projection: M=2048, N=32000, K=1024 -> (2048/32)*(32000/64)=32000 waves
  k_gemm<<<4000, 256, 0, stream>>>(HDEC, 1024, W2B, 1024, 0, m2b, out, 2048, 32000, 1024, 1);
}
